// IMISS_1382979469940
// MI455X (gfx1250) — matrix operand order repaired and checked by reading, not compiled, not run on hardware
//
#include <hip/hip_runtime.h>
#include <hip/hip_bf16.h>

typedef __attribute__((ext_vector_type(16))) _Float16 v16h;
typedef __attribute__((ext_vector_type(8)))  _Float16 h8;
typedef __attribute__((ext_vector_type(8)))  float    v8f;
typedef __attribute__((ext_vector_type(4)))  float    f4;

#define BB 16
#define HH 64
#define WW 64
#define DI 32
#define LL 4096
#define KK 4
#define NS 16

// A-matrix (16x32 f16) element->K mapping per ISA 7.12.2 (lane row = lane&15)
__device__ __forceinline__ int a_k(int e, int lane){
  return (e & 7) + ((e >> 3) << 4) + ((lane >> 4) << 3);
}
// B-matrix (32x16 f16) element->K mapping (lane col = lane&15)
__device__ __forceinline__ int b_k(int e, int lane){
  return (e & 7) + ((e >> 3) << 4) + ((lane >> 4) << 3);
}
// direction base ordering: kb=0 row-major l=h*64+w ; kb=1 transposed l=w*64+h
__device__ __forceinline__ int posmap(int kb, int p){
  return kb ? (((p & 63) << 6) | (p >> 6)) : p;
}
__device__ __forceinline__ float softplusf(float x){
  return (x > 20.f) ? x : log1pf(expf(x));
}
__device__ __forceinline__ float siluf(float x){
  return x / (1.f + expf(-x));
}

// ---------------- cc[b,c,k] = sum_hw q[b,hw,c]*s[hw,k] ------------------------------
// 8 waves/block split the K=4096 reduction; tiles staged through LDS with b128 loads.
__global__ void k_cc(const float* __restrict__ q, const float* __restrict__ s,
                     float* __restrict__ cc){
  __shared__ float qs[8][32][16];
  __shared__ float ss[8][32][16];
  __shared__ float red[8][32][8];
  int b = blockIdx.x;
  int tid = threadIdx.x;
  int wv = tid >> 5, lane = tid & 31;
  int m = lane & 15;
  v8f acc = {};
  for (int it = 0; it < 16; ++it){
    int hw0 = wv*512 + it*32;
    const f4* qp = (const f4*)(q + (size_t)(b*LL + hw0 + lane)*16);
    const f4* sp = (const f4*)(s + (size_t)(hw0 + lane)*16);
    f4* qd = (f4*)&qs[wv][lane][0];
    f4* sd = (f4*)&ss[wv][lane][0];
    qd[0]=qp[0]; qd[1]=qp[1]; qd[2]=qp[2]; qd[3]=qp[3];
    sd[0]=sp[0]; sd[1]=sp[1]; sd[2]=sp[2]; sd[3]=sp[3];
    __syncthreads();
    v16h a, bt;
#pragma unroll
    for (int e = 0; e < 16; ++e){
      a[e]  = (_Float16)qs[wv][a_k(e,lane)][m];
      bt[e] = (_Float16)ss[wv][b_k(e,lane)][m];
    }
    acc = __builtin_amdgcn_wmma_f32_16x16x32_f16(false, a, false, bt, (short)0, acc, false, false);
    __syncthreads();
  }
#pragma unroll
  for (int r = 0; r < 8; ++r) red[wv][lane][r] = acc[r];
  __syncthreads();
  if (wv == 0){
#pragma unroll
    for (int r = 0; r < 8; ++r){
      float t = 0.f;
      for (int w = 0; w < 8; ++w) t += red[w][lane][r];
      int M = r + ((lane >> 4) << 3);
      int N = lane & 15;
      cc[b*256 + M*16 + N] = t;
    }
  }
}

// ---------------- diagP[b,c] = softmax(cc[b,c,:] @ fc_w^T)[c] -----------------------
__global__ void k_diag(const float* __restrict__ cc, const float* __restrict__ fcw,
                       float* __restrict__ diagP){
  int t = blockIdx.x*blockDim.x + threadIdx.x;
  if (t >= BB*16) return;
  int b = t >> 4, c = t & 15;
  const float* row = cc + b*256 + c*16;
  float vals[16];
  float mx = -1e30f;
  for (int kp = 0; kp < 16; ++kp){
    float acc = 0.f;
    for (int k = 0; k < 16; ++k) acc += row[k]*fcw[kp*16 + k];
    vals[kp] = acc; mx = fmaxf(mx, acc);
  }
  float den = 0.f;
  for (int kp = 0; kp < 16; ++kp) den += expf(vals[kp]-mx);
  diagP[t] = expf(vals[c]-mx)/den;
}

// ---------------- in_proj: x_pre = q@Wq^T, s_pre = (s*diagP)@Ws^T (K=16 pad 32) -----
__global__ void k_inproj(const float* __restrict__ q, const float* __restrict__ s,
                         const float* __restrict__ diagP,
                         const float* __restrict__ wq, const float* __restrict__ ws,
                         float* __restrict__ x_pre, float* __restrict__ s_pre){
  int id = blockIdx.x;           // b*256 + tile
  int b = id >> 8, l0 = (id & 255) * 16;
  int lane = threadIdx.x;
  int m = lane & 15, lh = lane >> 4;
  // A fragments: valid K = lh*8 + 0..7 at elements 0..7 (contiguous); e>=8 -> pad 0
  const f4* qp = (const f4*)(q + (size_t)(b*LL + l0 + m)*16 + lh*8);
  const f4* sp = (const f4*)(s + (size_t)(l0 + m)*16 + lh*8);
  f4 qa0 = qp[0], qa1 = qp[1];
  f4 sa0 = sp[0], sa1 = sp[1];
  const f4* dp = (const f4*)(diagP + b*16 + lh*8);
  f4 d0 = dp[0], d1 = dp[1];
  v16h aq, as;
#pragma unroll
  for (int e = 0; e < 4; ++e){
    aq[e]    = (_Float16)qa0[e];
    aq[4+e]  = (_Float16)qa1[e];
    as[e]    = (_Float16)(sa0[e]*d0[e]);
    as[4+e]  = (_Float16)(sa1[e]*d1[e]);
    aq[8+e]  = (_Float16)0.f; aq[12+e] = (_Float16)0.f;
    as[8+e]  = (_Float16)0.f; as[12+e] = (_Float16)0.f;
  }
#pragma unroll
  for (int nt = 0; nt < 2; ++nt){
    int n = nt*16 + m;
    v16h bq, bs;
    {
      const f4* wqp = (const f4*)(wq + n*16 + lh*8);
      const f4* wsp = (const f4*)(ws + n*16 + lh*8);
      f4 w0=wqp[0], w1=wqp[1];
      f4 x0=wsp[0], x1=wsp[1];
#pragma unroll
      for (int e = 0; e < 4; ++e){
        bq[e]=(_Float16)w0[e]; bq[4+e]=(_Float16)w1[e];
        bq[8+e]=(_Float16)0.f; bq[12+e]=(_Float16)0.f;
        bs[e]=(_Float16)x0[e]; bs[4+e]=(_Float16)x1[e];
        bs[8+e]=(_Float16)0.f; bs[12+e]=(_Float16)0.f;
      }
    }
    v8f cq = {}, cs = {};
    cq = __builtin_amdgcn_wmma_f32_16x16x32_f16(false, aq, false, bq, (short)0, cq, false, false);
    cs = __builtin_amdgcn_wmma_f32_16x16x32_f16(false, as, false, bs, (short)0, cs, false, false);
#pragma unroll
    for (int r = 0; r < 8; ++r){
      int M = r + ((lane >> 4) << 3);
      int N = nt*16 + (lane & 15);
      x_pre[((size_t)b*DI + N)*LL + l0 + M] = cq[r];
      s_pre[((size_t)b*DI + N)*LL + l0 + M] = cs[r];
    }
  }
}

// ---------------- depthwise 3x3 + bias + SiLU on both buffers ------------------------
__global__ void k_dwconv(const float* __restrict__ x_pre, const float* __restrict__ s_pre,
                         const float* __restrict__ cw, const float* __restrict__ cb,
                         float* __restrict__ x_q, float* __restrict__ sc){
  size_t idx = (size_t)blockIdx.x*blockDim.x + threadIdx.x;
  const size_t N1 = (size_t)BB*DI*LL;
  if (idx >= 2*N1) return;
  const float* src = (idx < N1) ? x_pre : s_pre;
  float* dst = (idx < N1) ? x_q : sc;
  size_t i = (idx < N1) ? idx : idx - N1;
  int pos = (int)(i % LL);
  int ch  = (int)((i / LL) % DI);
  int b   = (int)(i / ((size_t)LL*DI));
  int h = pos >> 6, w = pos & 63;
  float acc = cb[ch];
  for (int ky = 0; ky < 3; ++ky){
    int y = h + ky - 1; if ((unsigned)y >= (unsigned)HH) continue;
    for (int kx = 0; kx < 3; ++kx){
      int x = w + kx - 1; if ((unsigned)x >= (unsigned)WW) continue;
      acc += src[((size_t)b*DI + ch)*LL + y*64 + x] * cw[ch*9 + ky*3 + kx];
    }
  }
  dst[i] = siluf(acc);
}

// ---------------- mask conv1 2->32 + relu + maxpool2 -> m1 NHWC (B,128,128,32) ------
__global__ void k_conv1(const float* __restrict__ smsk, const float* __restrict__ w1,
                        const float* __restrict__ b1, float* __restrict__ m1){
  size_t idx = (size_t)blockIdx.x*blockDim.x + threadIdx.x;
  const size_t N = (size_t)BB*128*128*32;
  if (idx >= N) return;
  int oc = (int)(idx & 31);
  int px = (int)((idx >> 5) & 127);
  int py = (int)((idx >> 12) & 127);
  int b  = (int)(idx >> 19);
  float best = -1e30f;
  for (int sy = 0; sy < 2; ++sy){
    for (int sx = 0; sx < 2; ++sx){
      int cy = py*2 + sy, cx = px*2 + sx;
      float acc = b1[oc];
      for (int ic = 0; ic < 2; ++ic){
        const float* im = smsk + ((size_t)(b*2 + ic))*256*256;
        const float* wk = w1 + (oc*2 + ic)*9;
        for (int ky = 0; ky < 3; ++ky){
          int y = cy + ky - 1; if ((unsigned)y >= 256u) continue;
          for (int kx = 0; kx < 3; ++kx){
            int x = cx + kx - 1; if ((unsigned)x >= 256u) continue;
            acc += im[y*256 + x] * wk[ky*3 + kx];
          }
        }
      }
      best = fmaxf(best, fmaxf(acc, 0.f));
    }
  }
  m1[idx] = best;   // ((b*128+py)*128+px)*32+oc
}

// ------- pack cm2 weights straight into WMMA B-fragment order, f16 -------------------
__global__ void k_w2pack(const float* __restrict__ w2, _Float16* __restrict__ w2pk){
  int idx = blockIdx.x*blockDim.x + threadIdx.x;
  if (idx >= 9*4*32*16) return;
  int e    = idx & 15;
  int lane = (idx >> 4) & 31;
  int nt   = (idx >> 9) & 3;
  int tap  = idx >> 11;
  int cout = nt*16 + (lane & 15);
  int cin  = (e & 7) + ((e >> 3) << 4) + ((lane >> 4) << 3);
  w2pk[idx] = (_Float16)w2[(cout*32 + cin)*9 + tap];
}

// ------- pack x_proj_weight into WMMA B-fragment order, f16 --------------------------
__global__ void k_xwpack(const float* __restrict__ xw, _Float16* __restrict__ xwpk){
  int idx = blockIdx.x*blockDim.x + threadIdx.x;
  if (idx >= 4*2*3*32*16) return;
  int e    = idx & 15;
  int lane = (idx >> 4) & 31;
  int rest = idx >> 9;
  int nt = rest % 3; rest /= 3;
  int ks = rest & 1;
  int k  = rest >> 1;
  int n  = nt*16 + (lane & 15);
  int kk = ks*32 + (e & 7) + ((e >> 3) << 4) + ((lane >> 4) << 3);
  float v = (n < 34) ? xw[((size_t)k*34 + n)*64 + kk] : 0.f;
  xwpk[idx] = (_Float16)v;
}

// ---------------- conv2 32->64 via WMMA im2col + relu + 2x2 pool -> mbuf (B,L,64) ---
__global__ void k_conv2(const float* __restrict__ m1, const _Float16* __restrict__ w2pk,
                        const float* __restrict__ b2, float* __restrict__ mbuf){
  int id = blockIdx.x;  // b*(64*16) + ry*16 + xt
  int xt = id & 15;
  int ry = (id >> 4) & 63;
  int b  = id >> 10;
  int y0 = ry*2, x0 = xt*8;
  int lane = threadIdx.x;
  int m = lane & 15, lh = lane >> 4;
  int pix_y = y0 + (m >> 3);
  int pix_x = x0 + (m & 7);
  int nlo = lane & 15;
  const v16h* wbv = (const v16h*)w2pk;
  v8f acc[4] = {{},{},{},{}};
#pragma unroll
  for (int tap = 0; tap < 9; ++tap){
    int ky = tap/3, kx = tap - ky*3;
    int iy = pix_y + ky - 1;
    int ix = pix_x + kx - 1;
    bool ok = ((unsigned)iy < 128u) && ((unsigned)ix < 128u);
    v16h a;
    if (ok){
      const f4* b4 = (const f4*)(m1 + (((size_t)b*128 + iy)*128 + ix)*32);
      f4 p0 = b4[lh*2], p1 = b4[lh*2+1];       // K = lh*8 .. lh*8+7
      f4 p2 = b4[4+lh*2], p3 = b4[5+lh*2];     // K = 16+lh*8 .. 16+lh*8+7
#pragma unroll
      for (int e = 0; e < 4; ++e){
        a[e]    = (_Float16)p0[e];
        a[4+e]  = (_Float16)p1[e];
        a[8+e]  = (_Float16)p2[e];
        a[12+e] = (_Float16)p3[e];
      }
    } else {
#pragma unroll
      for (int e = 0; e < 16; ++e) a[e] = (_Float16)0.f;
    }
#pragma unroll
    for (int nt = 0; nt < 4; ++nt){
      v16h bm = wbv[((tap*4 + nt) << 5) + lane];   // one 32B load
      acc[nt] = __builtin_amdgcn_wmma_f32_16x16x32_f16(false, a, false, bm, (short)0, acc[nt], false, false);
    }
  }
#pragma unroll
  for (int nt = 0; nt < 4; ++nt){
    float v[8], ov[8];
    float bias = b2[nt*16 + nlo];
#pragma unroll
    for (int r = 0; r < 8; ++r) v[r] = fmaxf(acc[nt][r] + bias, 0.f);
#pragma unroll
    for (int r = 0; r < 8; ++r) ov[r] = __shfl_xor(v[r], 16, 32); // other conv row
    if (lane < 16){
#pragma unroll
      for (int j = 0; j < 4; ++j){
        float pm = fmaxf(fmaxf(v[2*j], v[2*j+1]), fmaxf(ov[2*j], ov[2*j+1]));
        size_t pbase = (((size_t)b*4096) + ry*64 + xt*4 + j)*64;
        mbuf[pbase + nt*16 + nlo] = pm;
      }
    }
  }
}

// ---------------- pack interleaved xs (B,K,L,64) f16 ---------------------------------
__global__ void k_pack_xs(const float* __restrict__ x_q, const float* __restrict__ sc,
                          _Float16* __restrict__ xsA){
  int t = blockIdx.x*blockDim.x + threadIdx.x; // (b*4+k)*4096 + tau
  if (t >= BB*KK*LL) return;
  int tau = t & 4095;
  int k = (t >> 12) & 3;
  int b = t >> 14;
  int tau0 = (k >= 2) ? (4095 - tau) : tau;
  int kb = k & 1;
  int typ = tau0 & 1;
  _Float16 tmp[64];
#pragma unroll
  for (int c2 = 0; c2 < 64; ++c2){
    int j = c2 >> 1, i2 = c2 & 1;
    int p = (i2*4096 + tau0) >> 1;
    int pos = posmap(kb, p);
    float v = typ ? x_q[((size_t)b*DI + j)*LL + pos] : sc[((size_t)b*DI + j)*LL + pos];
    tmp[c2] = (_Float16)v;
  }
  v16h* dv = (v16h*)(xsA + (size_t)t*64);
  const v16h* tv = (const v16h*)tmp;
#pragma unroll
  for (int c = 0; c < 4; ++c) dv[c] = tv[c];
}

// ---------------- x_dbl (B,K,L,48) = xs^T @ x_proj_w^T via WMMA ----------------------
__global__ void k_xdbl(const _Float16* __restrict__ xsA, const _Float16* __restrict__ xwpk,
                       float* __restrict__ x_dbl){
  int id = blockIdx.x; // (b*4+k)*256 + tile
  int tile = id & 255;
  int k = (id >> 8) & 3;
  int b = id >> 10;
  int l0 = tile*16;
  int lane = threadIdx.x;
  int m = lane & 15, lh = lane >> 4;
  const _Float16* abase = xsA + (((size_t)(b*4 + k))*LL + l0)*64;
  const v16h* wbv = (const v16h*)xwpk;
  v8f acc[3] = {{},{},{}};
#pragma unroll
  for (int ks = 0; ks < 2; ++ks){
    const h8* ap = (const h8*)(abase + (size_t)m*64 + ks*32 + lh*8);
    h8 lo = ap[0];        // K = ks*32 + lh*8 + 0..7
    h8 hi = ap[2];        // K = ks*32 + 16 + lh*8 + 0..7
    v16h a;
#pragma unroll
    for (int e = 0; e < 8; ++e){ a[e] = lo[e]; a[8+e] = hi[e]; }
#pragma unroll
    for (int nt = 0; nt < 3; ++nt){
      v16h bm = wbv[((((k*2 + ks)*3 + nt)) << 5) + lane];
      acc[nt] = __builtin_amdgcn_wmma_f32_16x16x32_f16(false, a, false, bm, (short)0, acc[nt], false, false);
    }
  }
#pragma unroll
  for (int nt = 0; nt < 3; ++nt){
#pragma unroll
    for (int r = 0; r < 8; ++r){
      int M = r + ((lane >> 4) << 3);
      int N = nt*16 + (lane & 15);
      x_dbl[(((size_t)(b*4 + k))*LL + l0 + M)*48 + N] = acc[nt][r];
    }
  }
}

// ---------------- m_dbl (B,K,L,8): 5 outputs = s_msk_proj @ ms -----------------------
__global__ void k_mdbl(const float* __restrict__ mbuf, const float* __restrict__ msw,
                       float* __restrict__ m_dbl){
  int t = blockIdx.x*blockDim.x + threadIdx.x;
  if (t >= BB*KK*LL) return;
  int tau = t & 4095;
  int k = (t >> 12) & 3;
  int b = t >> 14;
  int tau0 = (k >= 2) ? (4095 - tau) : tau;
  int pos = posmap(k & 1, tau0);
  const f4* mr4 = (const f4*)(mbuf + ((size_t)b*4096 + pos)*64);
  float out[5] = {0.f,0.f,0.f,0.f,0.f};
  for (int dq = 0; dq < 16; ++dq){
    f4 mv = mr4[dq];
#pragma unroll
    for (int c = 0; c < 5; ++c){
      const float* wr = msw + ((size_t)k*5 + c)*64 + dq*4;
      out[c] += wr[0]*mv[0] + wr[1]*mv[1] + wr[2]*mv[2] + wr[3]*mv[3];
    }
  }
  float* dst = m_dbl + (size_t)t*8;
#pragma unroll
  for (int c = 0; c < 5; ++c) dst[c] = out[c];
}

__global__ void k_zero(float* __restrict__ p, size_t n){
  size_t i = (size_t)blockIdx.x*blockDim.x + threadIdx.x;
  if (i < n) p[i] = 0.f;
}

// ---------------- selective scan: 1 wave = 2 channels x 16 states --------------------
__global__ void k_scan(const float* __restrict__ x_dbl, const float* __restrict__ m_dbl,
                       const float* __restrict__ x_q, const float* __restrict__ sc,
                       const float* __restrict__ mbuf,
                       const float* __restrict__ A_logs, const float* __restrict__ Ds,
                       const float* __restrict__ dtw, const float* __restrict__ dtb,
                       const float* __restrict__ dt2w, const float* __restrict__ dt2b,
                       float* __restrict__ ycomb){
  int id = blockIdx.x;          // (b*4+k)*32 + pair
  int pr = id & 31;
  int k = (id >> 5) & 3;
  int b = id >> 7;
  int lane = threadIdx.x;
  int c2 = pr*2 + (lane >> 4);
  int n = lane & 15;
  int j = c2 >> 1, i2 = c2 & 1;
  int kb = k & 1;
  int gidx = k*64 + c2;
  float Aval = -expf(A_logs[gidx*16 + n]);
  float w0 = dtw[gidx*2 + 0], w1 = dtw[gidx*2 + 1];
  float bias = dtb[gidx];
  float v0 = dt2w[gidx*3 + 0], v1 = dt2w[gidx*3 + 1], v2 = dt2w[gidx*3 + 2];
  float bias2 = dt2b[gidx];
  float Dval = Ds[gidx];
  const float* xd  = x_dbl + ((size_t)(b*4 + k))*LL*48;
  const float* md  = m_dbl + ((size_t)(b*4 + k))*LL*8;
  const float* uqx = x_q + ((size_t)b*DI + j)*LL;
  const float* uqs = sc  + ((size_t)b*DI + j)*LL;
  const float* mb  = mbuf + (size_t)b*4096*64;
  float hp = 0.f, h1 = 0.f, h2 = 0.f, pf = 1.f;
  for (int tau = 0; tau < LL; ++tau){
    if ((tau & 7) == 0 && tau + 8 < LL){
      __builtin_prefetch(xd + (size_t)(tau + 8)*48, 0, 0);  // global_prefetch_b8
      __builtin_prefetch(md + (size_t)(tau + 8)*8, 0, 0);
    }
    int tau0 = (k >= 2) ? (4095 - tau) : tau;
    const float* xr = xd + (size_t)tau*48;
    float r0 = xr[0], r1 = xr[1];
    float dt = softplusf(r0*w0 + r1*w1 + bias);
    float Bn = xr[2 + n];
    float Cn = xr[18 + n];
    int p = (i2*4096 + tau0) >> 1;
    int pos = posmap(kb, p);
    float u = (tau0 & 1) ? uqx[pos] : uqs[pos];
    const float* mr = md + (size_t)tau*8;
    float dt2 = softplusf(mr[0]*v0 + mr[1]*v1 + mr[2]*v2 + bias2);
    float B2 = mr[3];
    int mpos = posmap(kb, tau0);
    float u2 = mb[(size_t)mpos*64 + c2];
    float a1 = expf(Aval*dt);
    h1 = a1*h1 + dt*u*Bn;           // feature-clock state
    float a2 = expf(Aval*dt2);
    h2 = a2*h2 + dt2*u2*B2;         // mask-clock state
    pf *= a1;                       // prefix decay (dt clock only)
    float h = h1 + h2 + pf*hp;
    float contrib = Cn*h;
    contrib += __shfl_xor(contrib, 8, 32);
    contrib += __shfl_xor(contrib, 4, 32);
    contrib += __shfl_xor(contrib, 2, 32);
    contrib += __shfl_xor(contrib, 1, 32);
    if ((tau0 & 1) && n == 0){      // only x-type steps survive deint
      atomicAdd(&ycomb[((size_t)b*DI + j)*LL + pos], contrib + Dval*u);
    }
    if ((tau & 7) == 7){ hp = h; h1 = 0.f; h2 = 0.f; pf = 1.f; }  // CHUNK=8
  }
}

// ---------------- layernorm + out_proj ----------------------------------------------
__global__ void k_final(const float* __restrict__ ycomb, const float* __restrict__ g,
                        const float* __restrict__ be, const float* __restrict__ wout,
                        float* __restrict__ out){
  int t = blockIdx.x*blockDim.x + threadIdx.x; // b*4096 + l
  if (t >= BB*LL) return;
  int l = t & 4095;
  int b = t >> 12;
  float v[32];
  float mu = 0.f;
  for (int c = 0; c < 32; ++c){ v[c] = ycomb[((size_t)b*DI + c)*LL + l]; mu += v[c]; }
  mu *= (1.f/32.f);
  float var = 0.f;
  for (int c = 0; c < 32; ++c){ float d = v[c]-mu; var += d*d; }
  var *= (1.f/32.f);
  float inv = rsqrtf(var + 1e-5f);
  float yn[32];
  for (int c = 0; c < 32; ++c) yn[c] = (v[c]-mu)*inv*g[c] + be[c];
  float* o = out + (size_t)t*16;
  for (int co = 0; co < 16; ++co){
    float acc = 0.f;
    for (int c = 0; c < 32; ++c) acc += wout[co*32 + c]*yn[c];
    o[co] = acc;
  }
}

extern "C" void kernel_launch(void* const* d_in, const int* in_sizes, int n_in,
                              void* d_out, int out_size, void* d_ws, size_t ws_size,
                              hipStream_t stream){
  const float* q    = (const float*)d_in[0];
  const float* s    = (const float*)d_in[1];
  const float* smsk = (const float*)d_in[2];
  const float* fcw  = (const float*)d_in[3];
  const float* ipw  = (const float*)d_in[4];
  const float* ipsw = (const float*)d_in[5];
  const float* cw   = (const float*)d_in[6];
  const float* cb   = (const float*)d_in[7];
  const float* xpw  = (const float*)d_in[8];
  const float* Alg  = (const float*)d_in[9];
  const float* Dsv  = (const float*)d_in[10];
  const float* dtw  = (const float*)d_in[11];
  const float* dtb  = (const float*)d_in[12];
  const float* onw  = (const float*)d_in[13];
  const float* onb  = (const float*)d_in[14];
  const float* opw  = (const float*)d_in[15];
  const float* w1   = (const float*)d_in[16];
  const float* b1   = (const float*)d_in[17];
  const float* w2   = (const float*)d_in[18];
  const float* b2   = (const float*)d_in[19];
  const float* msw  = (const float*)d_in[20];
  const float* dt2w = (const float*)d_in[21];
  const float* dt2b = (const float*)d_in[22];

  char* wp = (char*)d_ws;
  auto carve = [&](size_t bytes)->char*{
    char* p = wp; wp += (bytes + 255) & ~(size_t)255; return p;
  };
  float*    cc    = (float*)carve((size_t)16*256*4);
  float*    diagP = (float*)carve((size_t)256*4);
  float*    x_pre = (float*)carve((size_t)16*32*4096*4);
  float*    s_pre = (float*)carve((size_t)16*32*4096*4);
  float*    x_q   = (float*)carve((size_t)16*32*4096*4);
  float*    scb   = (float*)carve((size_t)16*32*4096*4);
  float*    m1    = (float*)carve((size_t)16*128*128*32*4);
  float*    mbuf  = (float*)carve((size_t)16*4096*64*4);
  _Float16* xsA   = (_Float16*)carve((size_t)16*4*4096*64*2);
  float*    x_dbl = (float*)carve((size_t)16*4*4096*48*4);
  float*    m_dbl = (float*)carve((size_t)16*4*4096*8*4);
  _Float16* w2pk  = (_Float16*)carve((size_t)9*4*32*16*2);
  _Float16* xwpk  = (_Float16*)carve((size_t)4*2*3*32*16*2);
  float*    ycomb = (float*)carve((size_t)16*32*4096*4);

  k_cc<<<16, 256, 0, stream>>>(q, s, cc);
  k_diag<<<1, 256, 0, stream>>>(cc, fcw, diagP);
  k_inproj<<<4096, 32, 0, stream>>>(q, s, diagP, ipw, ipsw, x_pre, s_pre);
  {
    size_t n = (size_t)2*16*32*4096;
    k_dwconv<<<(unsigned)((n+255)/256), 256, 0, stream>>>(x_pre, s_pre, cw, cb, x_q, scb);
  }
  {
    size_t n = (size_t)16*128*128*32;
    k_conv1<<<(unsigned)((n+255)/256), 256, 0, stream>>>(smsk, w1, b1, m1);
  }
  k_w2pack<<<(9*4*32*16+255)/256, 256, 0, stream>>>(w2, w2pk);
  k_xwpack<<<(4*2*3*32*16+255)/256, 256, 0, stream>>>(xpw, xwpk);
  k_conv2<<<16*64*16, 32, 0, stream>>>(m1, w2pk, b2, mbuf);
  k_pack_xs<<<(16*4*4096+255)/256, 256, 0, stream>>>(x_q, scb, xsA);
  k_xdbl<<<16*4*256, 32, 0, stream>>>(xsA, xwpk, x_dbl);
  k_mdbl<<<(16*4*4096+255)/256, 256, 0, stream>>>(mbuf, msw, m_dbl);
  {
    size_t n = (size_t)16*32*4096;
    k_zero<<<(unsigned)((n+255)/256), 256, 0, stream>>>(ycomb, n);
  }
  k_scan<<<16*4*32, 32, 0, stream>>>(x_dbl, m_dbl, x_q, scb, mbuf,
                                     Alg, Dsv, dtw, dtb, dt2w, dt2b, ycomb);
  k_final<<<(16*4096+255)/256, 256, 0, stream>>>(ycomb, onw, onb, opw, (float*)d_out);
}
